// GroupedQueryAttention_40269613367530
// MI455X (gfx1250) — matrix operand order repaired and checked by reading, not compiled, not run on hardware
//
#include <hip/hip_runtime.h>

// ---------------------------------------------------------------------------
// CDNA5 (gfx1250, wave32) WMMA: D = A(16x32 bf16) * B(32x16 bf16) + C(f32)
// ---------------------------------------------------------------------------
typedef __bf16 v16bf __attribute__((ext_vector_type(16)));
typedef float  v8f   __attribute__((ext_vector_type(8)));
typedef int    v4i   __attribute__((ext_vector_type(4)));

union FragU { v16bf v; uint4 q[2]; };

__device__ __forceinline__ v8f wmma_bf16(const FragU& a, const FragU& b, v8f c) {
  return __builtin_amdgcn_wmma_f32_16x16x32_bf16(false, a.v, false, b.v,
                                                 (short)0, c, false, false);
}

__device__ __forceinline__ unsigned short f2bf(float f) {
  union { float f; unsigned u; } c; c.f = f;
  unsigned u = c.u;
  unsigned r = u + 0x7FFFu + ((u >> 16) & 1u);   // round-to-nearest-even
  return (unsigned short)(r >> 16);
}

// ---------------------------------------------------------------------------
// Async global->LDS staging (CDNA5 ASYNCcnt path; signature probe-verified:
// (v4i AS1*, v4i AS3*, imm offset, imm cpol)), with sync fallback.
// ---------------------------------------------------------------------------
#if __has_builtin(__builtin_amdgcn_global_load_async_to_lds_b128)
#define HAVE_ASYNC 1
typedef __attribute__((address_space(1))) v4i gv4i;
typedef __attribute__((address_space(3))) v4i lv4i;
#else
#define HAVE_ASYNC 0
#endif

__device__ __forceinline__ void ld16_to_lds(const unsigned short* g,
                                            unsigned short* l) {
#if HAVE_ASYNC
  __builtin_amdgcn_global_load_async_to_lds_b128((gv4i*)(g), (lv4i*)(l), 0, 0);
#else
  *(uint4*)l = *(const uint4*)g;
#endif
}

__device__ __forceinline__ void wait_async() {
#if HAVE_ASYNC
#if __has_builtin(__builtin_amdgcn_s_wait_asynccnt)
  __builtin_amdgcn_s_wait_asynccnt(0);
#else
  asm volatile("s_wait_asynccnt 0" ::: "memory");
#endif
#endif
}

// ---------------------------------------------------------------------------
// fp32 -> bf16 conversion, 4 elements/thread (float4 in, ushort4 out)
// ---------------------------------------------------------------------------
__global__ __launch_bounds__(256) void cvt_f32_bf16(const float4* __restrict__ in,
                                                    ushort4* __restrict__ out,
                                                    int n4) {
  int i = blockIdx.x * 256 + threadIdx.x;
  if (i < n4) {
    float4 f = in[i];
    ushort4 o;
    o.x = f2bf(f.x); o.y = f2bf(f.y); o.z = f2bf(f.z); o.w = f2bf(f.w);
    out[i] = o;
  }
}

// ---------------------------------------------------------------------------
// GEMM:  C[M][N] = A[M][K](bf16) @ W[N][K]^T(bf16),  fp32 accumulate.
// 256 threads = 8 wave32s, block tile 256x128, wave tile 64x64, K-step 32.
// Double-buffered LDS, async staging, one barrier per K-step.
// ---------------------------------------------------------------------------
#define TSTR 40  // 32 -> 40 halfword row pad: conflict-free ds_load_b128

template <int OUT_BF16>
__global__ __launch_bounds__(256) void gemm_xwt(const unsigned short* __restrict__ A,
                                                const unsigned short* __restrict__ W,
                                                void* __restrict__ Cp,
                                                int M, int N, int K) {
  __shared__ __align__(16) unsigned short As[2][256 * TSTR];  // 40 KB
  __shared__ __align__(16) unsigned short Bs[2][128 * TSTR];  // 20 KB

  const int tid  = threadIdx.x;
  const int lane = tid & 31, wave = tid >> 5;
  const int g = lane >> 4, ln = lane & 15;
  const int wm = wave >> 1, wn = wave & 1;  // 4x2 wave grid, 64x64 each
  const int mBase = blockIdx.y * 256, nBase = blockIdx.x * 128;

  v8f acc[4][4] = {};

  auto stage = [&](int buf, int k0) {
#pragma unroll
    for (int i = 0; i < 4; ++i) {                 // A tile: 256 rows x 32
      int li = tid + i * 256;
      int row = li >> 2, c = (li & 3) << 3;
      ld16_to_lds(&A[(size_t)(mBase + row) * K + k0 + c],
                  &As[buf][row * TSTR + c]);
    }
#pragma unroll
    for (int i = 0; i < 2; ++i) {                 // B tile: 128 rows x 32
      int li = tid + i * 256;
      int row = li >> 2, c = (li & 3) << 3;
      ld16_to_lds(&W[(size_t)(nBase + row) * K + k0 + c],
                  &Bs[buf][row * TSTR + c]);
    }
  };

  const int kIters = K >> 5;
  stage(0, 0);
  wait_async();

  for (int it = 0; it < kIters; ++it) {
    __syncthreads();  // buf(it&1) fully written; buf((it+1)&1) free for reuse
    if (it + 1 < kIters) stage((it + 1) & 1, (it + 1) << 5);

    const unsigned short* Ab = As[it & 1];
    const unsigned short* Bb = Bs[it & 1];

    // A fragments (ISA 16-bit A layout: VGPR0-3 = K[g*8..+7], VGPR4-7 = +16)
    FragU af[4];
#pragma unroll
    for (int s = 0; s < 4; ++s) {
      int base = (wm * 64 + s * 16 + ln) * TSTR;
      af[s].q[0] = *(const uint4*)&Ab[base + g * 8];
      af[s].q[1] = *(const uint4*)&Ab[base + 16 + g * 8];
    }
    // B fragments (lane = N, K = g*8..+7 and 16+g*8..+7, matching A)
    FragU bf4[4];
#pragma unroll
    for (int t = 0; t < 4; ++t) {
      int base = (wn * 64 + t * 16 + ln) * TSTR;
      bf4[t].q[0] = *(const uint4*)&Bb[base + g * 8];
      bf4[t].q[1] = *(const uint4*)&Bb[base + 16 + g * 8];
    }
#pragma unroll
    for (int s = 0; s < 4; ++s)
#pragma unroll
      for (int t = 0; t < 4; ++t)
        acc[s][t] = wmma_bf16(af[s], bf4[t], acc[s][t]);

    wait_async();  // own prefetch into buf((it+1)&1) done before next barrier
  }

  // C layout: VGPR r holds (M = r + 8*g, N = ln)
#pragma unroll
  for (int s = 0; s < 4; ++s)
#pragma unroll
    for (int t = 0; t < 4; ++t)
#pragma unroll
      for (int r = 0; r < 8; ++r) {
        int row = mBase + wm * 64 + s * 16 + r + 8 * g;
        int col = nBase + wn * 64 + t * 16 + ln;
        float v = acc[s][t][r];
        if (OUT_BF16)
          ((unsigned short*)Cp)[(size_t)row * N + col] = f2bf(v);
        else
          ((float*)Cp)[(size_t)row * N + col] = v;
      }
}

// ---------------------------------------------------------------------------
// Fused causal GQA flash attention (bf16 Q/K/V, fp32 accumulate, bf16 ctx out)
// grid = (N/64, B*H), block = 128 threads (4 wave32s), 16 query rows per wave.
// ---------------------------------------------------------------------------
__global__ __launch_bounds__(128) void gqa_attn(const unsigned short* __restrict__ Q,
                                                const unsigned short* __restrict__ Kg,
                                                const unsigned short* __restrict__ Vg,
                                                unsigned short* __restrict__ Ctx) {
  __shared__ __align__(16) unsigned short Ksh[32 * 72];  // [key][hd], pad 64->72
  __shared__ __align__(16) unsigned short Vt[64 * 40];   // [hd][key], pad 32->40
  __shared__ __align__(16) unsigned short Psh[4][16 * 40];

  const int tid  = threadIdx.x;
  const int lane = tid & 31, wave = tid >> 5;
  const int g = lane >> 4, ln = lane & 15;

  const int bh = blockIdx.y;
  const int b = bh >> 5, h = bh & 31;  // H = 32
  const int kvh = h >> 2;              // G = 4 query heads per kv head
  const int rowBase = blockIdx.x * 64;
  const size_t bN = (size_t)b * 2048;

  // Q fragments: 16 rows x 64 (HD) = 2 K-steps, kept in registers
  FragU qf[2];
  {
    const int row = rowBase + wave * 16 + ln;
    const size_t qrow = (bN + row) * 2048 + (size_t)h * 64;
#pragma unroll
    for (int s = 0; s < 2; ++s) {
      qf[s].q[0] = *(const uint4*)&Q[qrow + s * 32 + g * 8];
      qf[s].q[1] = *(const uint4*)&Q[qrow + s * 32 + 16 + g * 8];
    }
  }

  float m_run[8], l_run[8];
  v8f o[4] = {};
#pragma unroll
  for (int r = 0; r < 8; ++r) { m_run[r] = -1e30f; l_run[r] = 0.0f; }

  const int nblk = 2 * blockIdx.x + 2;  // key blocks of 32 up to rowBase+63
  for (int j = 0; j < nblk; ++j) {
    __syncthreads();
    // K block -> Ksh [32][72] (hd contiguous) via async LDS path
#pragma unroll
    for (int i = 0; i < 2; ++i) {
      int li = tid + i * 128;
      int key = li >> 3, c = (li & 7) << 3;
      ld16_to_lds(&Kg[(bN + j * 32 + key) * 512 + (size_t)kvh * 64 + c],
                  &Ksh[key * 72 + c]);
    }
    // V block transposed -> Vt[hd][key]: coalesced uint2 load phase into
    // independent registers, then 16 ds_store_b16 scatter (no per-pair waits)
    {
      uint2 vl[4];
      int keyv[4], hdq[4];
#pragma unroll
      for (int i = 0; i < 4; ++i) {
        int idx4 = i * 128 + tid;
        keyv[i] = idx4 >> 4;
        hdq[i]  = (idx4 & 15) << 2;
        vl[i] = *(const uint2*)&Vg[(bN + j * 32 + keyv[i]) * 512 +
                                   (size_t)kvh * 64 + hdq[i]];
      }
#pragma unroll
      for (int i = 0; i < 4; ++i) {
        int base = hdq[i] * 40 + keyv[i];
        Vt[base]       = (unsigned short)(vl[i].x & 0xFFFF);
        Vt[base + 40]  = (unsigned short)(vl[i].x >> 16);
        Vt[base + 80]  = (unsigned short)(vl[i].y & 0xFFFF);
        Vt[base + 120] = (unsigned short)(vl[i].y >> 16);
      }
    }
    wait_async();
    __syncthreads();

    // S[16][32] = Q @ K^T : B[k=hd][n=key], per-lane reads contiguous in Ksh row
    v8f sc[2] = {};
#pragma unroll
    for (int t = 0; t < 2; ++t)
#pragma unroll
      for (int s = 0; s < 2; ++s) {
        FragU kf;
        int base = (t * 16 + ln) * 72 + s * 32;
        kf.q[0] = *(const uint4*)&Ksh[base + g * 8];
        kf.q[1] = *(const uint4*)&Ksh[base + 16 + g * 8];
        sc[t] = wmma_bf16(qf[s], kf, sc[t]);
      }

    // scale (1/sqrt(64)) + causal mask; online softmax (row spans 16 lanes/half)
    float s0[8], s1[8], cs[8];
#pragma unroll
    for (int r = 0; r < 8; ++r) {
      int row = rowBase + wave * 16 + r + 8 * g;
      int c0 = j * 32 + ln;
      s0[r] = (c0 <= row)      ? sc[0][r] * 0.125f : -1e30f;
      s1[r] = (c0 + 16 <= row) ? sc[1][r] * 0.125f : -1e30f;
    }
#pragma unroll
    for (int r = 0; r < 8; ++r) {
      float mx = fmaxf(s0[r], s1[r]);
#pragma unroll
      for (int d = 1; d < 16; d <<= 1) mx = fmaxf(mx, __shfl_xor(mx, d, 32));
      float mnew = fmaxf(m_run[r], mx);
      cs[r] = __expf(m_run[r] - mnew);
      m_run[r] = mnew;
      float p0 = __expf(s0[r] - mnew);
      float p1 = __expf(s1[r] - mnew);
      s0[r] = p0; s1[r] = p1;
      float ls = p0 + p1;
#pragma unroll
      for (int d = 1; d < 16; d <<= 1) ls += __shfl_xor(ls, d, 32);
      l_run[r] = l_run[r] * cs[r] + ls;
    }

    // rescale running O
#pragma unroll
    for (int t = 0; t < 4; ++t)
#pragma unroll
      for (int r = 0; r < 8; ++r) o[t][r] *= cs[r];

    // P (C-layout fp32) -> wave-private LDS (bf16) -> A-layout fragment
    unsigned short* P = Psh[wave];
#pragma unroll
    for (int r = 0; r < 8; ++r) {
      int pr = (r + 8 * g) * 40;
      P[pr + ln]      = f2bf(s0[r]);
      P[pr + 16 + ln] = f2bf(s1[r]);
    }
    FragU pf;
    {
      int base = ln * 40;
      pf.q[0] = *(const uint4*)&P[base + g * 8];
      pf.q[1] = *(const uint4*)&P[base + 16 + g * 8];
    }

    // O[16][64] += P[16x32] @ V[32x64] : one K=32 step, 4 N-subtiles
#pragma unroll
    for (int t = 0; t < 4; ++t) {
      FragU vf;
      int base = (t * 16 + ln) * 40;
      vf.q[0] = *(const uint4*)&Vt[base + g * 8];
      vf.q[1] = *(const uint4*)&Vt[base + 16 + g * 8];
      o[t] = wmma_bf16(pf, vf, o[t]);
    }
  }

  // normalize and store ctx as bf16 at [b, row, h*64 + n] ([B,N,EMBED] flat)
#pragma unroll
  for (int r = 0; r < 8; ++r) {
    float inv = 1.0f / l_run[r];
    int row = rowBase + wave * 16 + r + 8 * g;
    size_t base = (bN + row) * 2048 + (size_t)h * 64;
#pragma unroll
    for (int t = 0; t < 4; ++t)
      Ctx[base + t * 16 + ln] = f2bf(o[t][r] * inv);
  }
}

// ---------------------------------------------------------------------------
// Host: convert -> 3 projections -> fused attention -> output projection
// ---------------------------------------------------------------------------
extern "C" void kernel_launch(void* const* d_in, const int* in_sizes, int n_in,
                              void* d_out, int out_size, void* d_ws, size_t ws_size,
                              hipStream_t stream) {
  const float* x  = (const float*)d_in[0];
  const float* Wq = (const float*)d_in[1];
  const float* Wk = (const float*)d_in[2];
  const float* Wv = (const float*)d_in[3];
  const float* Wo = (const float*)d_in[4];

  char* ws = (char*)d_ws;
  size_t off = 0;
  auto take = [&](size_t elems) {
    unsigned short* p = (unsigned short*)(ws + off);
    off = (off + elems * 2 + 255) & ~(size_t)255;
    return p;
  };
  unsigned short* xb  = take(8388608);  // [2,2048,2048]
  unsigned short* Wqb = take(4194304);  // [2048,2048]
  unsigned short* Wkb = take(1048576);  // [512,2048]
  unsigned short* Wvb = take(1048576);
  unsigned short* Wob = take(4194304);
  unsigned short* Qb  = take(8388608);  // [4096,2048]
  unsigned short* Kb  = take(2097152);  // [4096,512]
  unsigned short* Vb  = take(2097152);
  unsigned short* Cx  = take(8388608);  // [4096,2048]

  auto cvt = [&](const float* i, unsigned short* o, int n) {
    int n4 = n >> 2;
    cvt_f32_bf16<<<(n4 + 255) / 256, 256, 0, stream>>>(
        (const float4*)i, (ushort4*)o, n4);
  };
  cvt(x,  xb,  8388608);
  cvt(Wq, Wqb, 4194304);
  cvt(Wk, Wkb, 1048576);
  cvt(Wv, Wvb, 1048576);
  cvt(Wo, Wob, 4194304);

  // Q/K/V projections: [4096,2048] @ W^T  (block tile 256x128)
  gemm_xwt<1><<<dim3(16, 16), 256, 0, stream>>>(xb, Wqb, Qb, 4096, 2048, 2048);
  gemm_xwt<1><<<dim3(4,  16), 256, 0, stream>>>(xb, Wkb, Kb, 4096, 512,  2048);
  gemm_xwt<1><<<dim3(4,  16), 256, 0, stream>>>(xb, Wvb, Vb, 4096, 512,  2048);

  // fused causal GQA attention: grid (N/64 row blocks, B*H heads)
  gqa_attn<<<dim3(32, 64), 128, 0, stream>>>(Qb, Kb, Vb, Cx);

  // output projection -> fp32 d_out
  gemm_xwt<0><<<dim3(16, 16), 256, 0, stream>>>(Cx, Wob, (float*)d_out,
                                                4096, 2048, 2048);
}
